// JTMPN_35158602285572
// MI455X (gfx1250) — matrix operand order repaired and checked by reading, not compiled, not run on hardware
//
#include <hip/hip_runtime.h>

// ---- problem constants (from reference) ----
#define HIDDEN   128
#define ATOM_FDIM 35
#define BOND_FDIM 5
#define BFD      (ATOM_FDIM + BOND_FDIM)   // 40
#define WO_K     (ATOM_FDIM + HIDDEN)      // 163
#define N_ATOMS  50000
#define N_BONDS  100000
#define N_TREE   20000
#define N_MSG    (N_TREE + N_BONDS)        // 120000
#define N_MOLS   500
#define MAX_NB   15
#define KPAD     192                        // 64 (fatoms padded) + 128 (nei)

typedef __attribute__((ext_vector_type(16))) __bf16 v16bf;
typedef __attribute__((ext_vector_type(8)))  __bf16 v8bf;
typedef __attribute__((ext_vector_type(4)))  __bf16 v4bf;
typedef __attribute__((ext_vector_type(8)))  float  v8f;

// A fragment (16x32 bf16): lane row M = lane&15; half-group g picks interleaved
// K octets: e0..7 -> K = 32*ks + 8g + e, e8..15 -> K = 32*ks + 16 + 8g + (e-8).
__device__ __forceinline__ v16bf load_afrag(const __bf16* p) {
    v8bf lo = *(const v8bf*)p;          // one ds_load_b128
    v8bf hi = *(const v8bf*)(p + 16);   // one ds_load_b128
    return __builtin_shufflevector(lo, hi, 0,1,2,3,4,5,6,7,8,9,10,11,12,13,14,15);
}

// ---------------------------------------------------------------------------
// K1b: tree_message (f32) -> bf16 rows [0, N_TREE) of both message buffers
// ---------------------------------------------------------------------------
__global__ void k_tree(const float* __restrict__ tm,
                       __bf16* __restrict__ msg0, __bf16* __restrict__ msg1) {
    int i = blockIdx.x * blockDim.x + threadIdx.x;
    if (i < N_TREE * HIDDEN) {
        __bf16 v = (__bf16)tm[i];
        msg0[i] = v;
        msg1[i] = v;
    }
}

// ---------------------------------------------------------------------------
// K1: binput = fbonds @ W_i.T  (K=40, tiny);  msg0 graph rows = bf16(relu)
// blockDim = 128 (thread == output channel), 64 bonds per block
// ---------------------------------------------------------------------------
__global__ void k_wi(const float* __restrict__ fbonds, const float* __restrict__ W_i,
                     float* __restrict__ binput, __bf16* __restrict__ msg0) {
    __shared__ float Wi[HIDDEN * BFD];            // 20 KB
    for (int i = threadIdx.x; i < HIDDEN * BFD; i += blockDim.x) Wi[i] = W_i[i];
    __syncthreads();
    int h  = threadIdx.x;
    int b0 = blockIdx.x * 64;
    for (int bb = 0; bb < 64; ++bb) {
        int b = b0 + bb;
        if (b >= N_BONDS) break;
        const float* fb = fbonds + (size_t)b * BFD;
        float s = 0.f;
        #pragma unroll
        for (int k = 0; k < BFD; ++k) s += fb[k] * Wi[h * BFD + k];
        binput[(size_t)b * HIDDEN + h] = s;
        msg0[(size_t)(N_TREE + b) * HIDDEN + h] = (__bf16)fmaxf(s, 0.f);
    }
}

// ---------------------------------------------------------------------------
// K2: one message-passing step.
// blockDim = 128 (4 waves); each wave owns a 16-bond tile.
//   gather-sum 15 nbr rows (bf16, fp32 acc) -> LDS bf16 A-stage
//   WMMA bf16 16x16x32: [16 x 128] @ W_h.T -> +binput, ReLU -> msg_out bf16
// Per k-step: batch all 8 B-frag LDS loads into distinct regs, then 8 WMMAs
// back-to-back (lets the scheduler use partial s_wait_dscnt, keeps XDL busy).
// ---------------------------------------------------------------------------
__global__ void k_mp(const __bf16* __restrict__ msg_in, const float* __restrict__ W_h,
                     const float* __restrict__ binput, const int* __restrict__ bgraph,
                     __bf16* __restrict__ msg_out) {
    __shared__ __bf16 Wh[HIDDEN * HIDDEN];        // 32 KB, B as [n][k] = W_h row-major
    __shared__ __bf16 Ast[4][16 * HIDDEN];        // 16 KB
    int tid = threadIdx.x;
    for (int i = tid; i < HIDDEN * HIDDEN; i += 128) Wh[i] = (__bf16)W_h[i];
    __syncthreads();

    int w = tid >> 5, lane = tid & 31;
    int g = lane >> 4, nlo = lane & 15;
    int tile = (blockIdx.x * 4 + w) * 16;
    __bf16* As = Ast[w];

    // ---- gather phase: lane handles 4 consecutive hidden cols ----
    for (int m = 0; m < 16; ++m) {
        int bond = tile + m;
        float a0 = 0.f, a1 = 0.f, a2 = 0.f, a3 = 0.f;
        if (bond < N_BONDS) {
            const int* nb = bgraph + (size_t)bond * MAX_NB;
            #pragma unroll
            for (int j = 0; j < MAX_NB; ++j) {
                int row = nb[j];
                v4bf v = *(const v4bf*)(msg_in + (size_t)row * HIDDEN + lane * 4);
                a0 += (float)v.x; a1 += (float)v.y; a2 += (float)v.z; a3 += (float)v.w;
            }
        }
        v4bf o; o.x = (__bf16)a0; o.y = (__bf16)a1; o.z = (__bf16)a2; o.w = (__bf16)a3;
        *(v4bf*)(As + m * HIDDEN + lane * 4) = o;
    }
    __syncthreads();

    // ---- WMMA phase: M=16, N=128 (8 tiles), K=128 (4 steps) ----
    v8f acc[8];
    v8f zero = {};
    #pragma unroll
    for (int nt = 0; nt < 8; ++nt) acc[nt] = zero;

    const __bf16* arow = As + nlo * HIDDEN + g * 8;
    const __bf16* brow = Wh + nlo * HIDDEN + g * 8;

    #pragma unroll
    for (int ks = 0; ks < 4; ++ks) {
        v16bf a = load_afrag(arow + ks * 32);
        v16bf b[8];
        #pragma unroll
        for (int nt = 0; nt < 8; ++nt)
            b[nt] = load_afrag(brow + nt * 16 * HIDDEN + ks * 32);
        #pragma unroll
        for (int nt = 0; nt < 8; ++nt)
            acc[nt] = __builtin_amdgcn_wmma_f32_16x16x32_bf16(
                false, a, false, b[nt], (short)0, acc[nt], false, false);
    }

    // ---- epilogue: + binput, ReLU, store bf16 (C layout: M = r + 8*g, N = nlo) ----
    #pragma unroll
    for (int nt = 0; nt < 8; ++nt) {
        int col = nt * 16 + nlo;
        #pragma unroll
        for (int r = 0; r < 8; ++r) {
            int bond = tile + r + 8 * g;
            if (bond < N_BONDS) {
                float v = binput[(size_t)bond * HIDDEN + col] + acc[nt][r];
                msg_out[(size_t)(N_TREE + bond) * HIDDEN + col] = (__bf16)fmaxf(v, 0.f);
            }
        }
    }
}

// ---------------------------------------------------------------------------
// K3: atom readout + fused mean pooling.
// A = [fatoms(35) pad->64 | gathered nei(128)] (K=192), B = padded W_o bf16.
// relu(A @ B.T + b_o) / scope_len  atomically added into d_out[mol].
// blockDim = 64 (2 waves); each wave owns a 16-atom tile.
// ---------------------------------------------------------------------------
__global__ void k_atom(const __bf16* __restrict__ msg, const float* __restrict__ fatoms,
                       const float* __restrict__ W_o, const float* __restrict__ b_o,
                       const int* __restrict__ agraph, const int* __restrict__ mol_idx,
                       const int* __restrict__ scope_len, float* __restrict__ out) {
    __shared__ __bf16 WoP[HIDDEN * KPAD];         // 48 KB
    __shared__ __bf16 Ast[2][16 * KPAD];          // 12 KB
    int tid = threadIdx.x;                         // blockDim 64
    // padded-W_o fill, no integer division: rows outer, cols strided by 64
    for (int n = 0; n < HIDDEN; ++n) {
        const float* wr = W_o + (size_t)n * WO_K;
        __bf16* dr = WoP + n * KPAD;
        for (int kk = tid; kk < KPAD; kk += 64) {
            float v = 0.f;
            if (kk < ATOM_FDIM)  v = wr[kk];
            else if (kk >= 64)   v = wr[kk - 64 + ATOM_FDIM];
            dr[kk] = (__bf16)v;
        }
    }
    __syncthreads();

    int w = tid >> 5, lane = tid & 31;
    int g = lane >> 4, nlo = lane & 15;
    int tile = (blockIdx.x * 2 + w) * 16;
    __bf16* As = Ast[w];

    for (int m = 0; m < 16; ++m) {
        int atom = tile + m;
        bool valid = atom < N_ATOMS;
        // fatoms / pad region kk in [0,64): lanes 0..15
        if (lane < 16) {
            v4bf o;
            #pragma unroll
            for (int j = 0; j < 4; ++j) {
                int kk = lane * 4 + j;
                float v = (valid && kk < ATOM_FDIM) ? fatoms[(size_t)atom * ATOM_FDIM + kk] : 0.f;
                o[j] = (__bf16)v;
            }
            *(v4bf*)(As + m * KPAD + lane * 4) = o;
        }
        // nei region kk in [64,192): all 32 lanes, 4 cols each
        float a0 = 0.f, a1 = 0.f, a2 = 0.f, a3 = 0.f;
        if (valid) {
            const int* nb = agraph + (size_t)atom * MAX_NB;
            #pragma unroll
            for (int j = 0; j < MAX_NB; ++j) {
                int row = nb[j];
                v4bf v = *(const v4bf*)(msg + (size_t)row * HIDDEN + lane * 4);
                a0 += (float)v.x; a1 += (float)v.y; a2 += (float)v.z; a3 += (float)v.w;
            }
        }
        v4bf o; o.x = (__bf16)a0; o.y = (__bf16)a1; o.z = (__bf16)a2; o.w = (__bf16)a3;
        *(v4bf*)(As + m * KPAD + 64 + lane * 4) = o;
    }
    __syncthreads();

    v8f acc[8];
    v8f zero = {};
    #pragma unroll
    for (int nt = 0; nt < 8; ++nt) acc[nt] = zero;

    const __bf16* arow = As + nlo * KPAD + g * 8;
    const __bf16* brow = WoP + nlo * KPAD + g * 8;

    #pragma unroll
    for (int ks = 0; ks < 6; ++ks) {              // K = 192 = 6 x 32
        v16bf a = load_afrag(arow + ks * 32);
        v16bf b[8];
        #pragma unroll
        for (int nt = 0; nt < 8; ++nt)
            b[nt] = load_afrag(brow + nt * 16 * KPAD + ks * 32);
        #pragma unroll
        for (int nt = 0; nt < 8; ++nt)
            acc[nt] = __builtin_amdgcn_wmma_f32_16x16x32_bf16(
                false, a, false, b[nt], (short)0, acc[nt], false, false);
    }

    #pragma unroll
    for (int nt = 0; nt < 8; ++nt) {
        int col = nt * 16 + nlo;
        float bo = b_o[col];
        #pragma unroll
        for (int r = 0; r < 8; ++r) {
            int atom = tile + r + 8 * g;
            if (atom < N_ATOMS) {
                float v = fmaxf(acc[nt][r] + bo, 0.f);
                int mol = mol_idx[atom];
                atomicAdd(out + (size_t)mol * HIDDEN + col, v / (float)scope_len[mol]);
            }
        }
    }
}

// ---------------------------------------------------------------------------
extern "C" void kernel_launch(void* const* d_in, const int* in_sizes, int n_in,
                              void* d_out, int out_size, void* d_ws, size_t ws_size,
                              hipStream_t stream) {
    const float* fatoms   = (const float*)d_in[0];
    const float* fbonds   = (const float*)d_in[1];
    const float* tree_msg = (const float*)d_in[2];
    const int*   agraph   = (const int*)d_in[3];
    const int*   bgraph   = (const int*)d_in[4];
    const int*   mol_idx  = (const int*)d_in[5];
    const int*   scope_ln = (const int*)d_in[6];
    const float* W_i      = (const float*)d_in[7];
    const float* W_h      = (const float*)d_in[8];
    const float* W_o      = (const float*)d_in[9];
    const float* b_o      = (const float*)d_in[10];
    float* out = (float*)d_out;

    // workspace layout (bf16 message double-buffer + fp32 binput) ~113 MB
    const size_t MSG_BYTES = (size_t)N_MSG * HIDDEN * sizeof(__bf16);   // 30,720,000
    char* ws = (char*)d_ws;
    __bf16* msg0   = (__bf16*)(ws);
    __bf16* msg1   = (__bf16*)(ws + MSG_BYTES);
    float*  binput = (float*)(ws + 2 * MSG_BYTES);

    hipMemsetAsync(out, 0, (size_t)N_MOLS * HIDDEN * sizeof(float), stream);

    // tree rows into both buffers
    {
        int n = N_TREE * HIDDEN;
        k_tree<<<(n + 255) / 256, 256, 0, stream>>>(tree_msg, msg0, msg1);
    }
    // initial bond messages
    k_wi<<<(N_BONDS + 63) / 64, 128, 0, stream>>>(fbonds, W_i, binput, msg0);

    // 3 message-passing steps, ping-pong (final message in msg1)
    const int mp_grid = (N_BONDS + 63) / 64;      // 4 waves x 16 bonds per block
    k_mp<<<mp_grid, 128, 0, stream>>>(msg0, W_h, binput, bgraph, msg1);
    k_mp<<<mp_grid, 128, 0, stream>>>(msg1, W_h, binput, bgraph, msg0);
    k_mp<<<mp_grid, 128, 0, stream>>>(msg0, W_h, binput, bgraph, msg1);

    // atom readout + fused mean pooling
    const int at_grid = (N_ATOMS + 31) / 32;      // 2 waves x 16 atoms per block
    k_atom<<<at_grid, 64, 0, stream>>>(msg1, fatoms, W_o, b_o, agraph,
                                       mol_idx, scope_ln, out);
}
